// MoELanguageModel_81630148428168
// MI455X (gfx1250) — matrix operand order repaired and checked by reading, not compiled, not run on hardware
//
#include <hip/hip_runtime.h>
#include <hip/hip_bf16.h>
#include <math.h>

// ---------------- types ----------------
typedef __attribute__((ext_vector_type(16))) _Float16 v16h;
typedef __attribute__((ext_vector_type(8)))  _Float16 v8h;
typedef __attribute__((ext_vector_type(8)))  float    v8f;
typedef __attribute__((ext_vector_type(4)))  float    v4f;

union FragAB { v16h v; v8h h[2]; };

// ---------------- model dims (fixed by reference) ----------------
#define BDIM  2
#define SDIM  1024
#define DDIM  1024
#define HDIM  4096
#define EEXP  8
#define VVOC  32000
#define NHEAD 16
#define HDHD  64
#define NLAY  2

// =====================================================================
// Generic WMMA GEMM:  C[z] = epi( A[z](MxK) * B[z]^T or B[z](KxN) + bias )
//   A row-major MxK (lda);  BT=true: B row-major NxK (x@W.T layout)
//                           BT=false: B row-major KxN
//   EPI: 0=none, 1=+bias[col], 2=+bias[col] then exact GELU
//   z-batch: off = (z/zmod)*OffD + (z%zmod)*OffM  per operand
//   Wave tile = (BM/WROWS) x (BN/WCOLS); block = WROWS*WCOLS waves.
// =====================================================================
template<int BM, int BN, int BK, int WROWS, int WCOLS, bool BT, int EPI>
__global__ __launch_bounds__(WROWS * WCOLS * 32) void gemm_f16_wmma(
    const float* __restrict__ Abase, const float* __restrict__ Bbase,
    float* __restrict__ Cbase, const float* __restrict__ bias,
    int M, int N, int K, int lda, int ldb, int ldc,
    int zmod, long long aOffD, long long aOffM,
    long long bOffD, long long bOffM, long long cOffD, long long cOffM)
{
    constexpr int THREADS = WROWS * WCOLS * 32;
    constexpr int WM = BM / WROWS;
    constexpr int WN = BN / WCOLS;
    constexpr int MI = WM / 16;
    constexpr int NI = WN / 16;
    constexpr int LDK = BK + 8;                  // halfs; pad vs bank conflicts

    __shared__ __align__(16) _Float16 As[BM * LDK];
    __shared__ __align__(16) _Float16 Bs[BN * LDK];

    const int tid  = threadIdx.x;
    const int wave = tid >> 5;
    const int lane = tid & 31;
    const int g    = lane >> 4;                  // lane half-group
    const int l16  = lane & 15;
    const int wrow = wave % WROWS;
    const int wcol = wave / WROWS;

    const int z = blockIdx.z;
    const float* A = Abase + (long long)(z / zmod) * aOffD + (long long)(z % zmod) * aOffM;
    const float* B = Bbase + (long long)(z / zmod) * bOffD + (long long)(z % zmod) * bOffM;
    float*       C = Cbase + (long long)(z / zmod) * cOffD + (long long)(z % zmod) * cOffM;

    const int m0 = blockIdx.y * BM;
    const int n0 = blockIdx.x * BN;

    v8f acc[MI][NI] = {};

    for (int kt = 0; kt < K; kt += BK) {
        // ---- stage A tile (BM x BK), fp32 -> f16 ----
        {
            constexpr int RP = THREADS / (BK / 4);   // rows per pass
            const int lr = tid / (BK / 4);
            const int lc = (tid % (BK / 4)) * 4;
            #pragma unroll 4
            for (int r = 0; r < BM; r += RP) {
                const float* src = A + (long long)(m0 + r + lr) * lda + kt + lc;
                __builtin_prefetch(src + BK, 0, 1);  // speculative; dropped if OOB
                v4f a = *(const v4f*)src;
                _Float16* dst = &As[(r + lr) * LDK + lc];
                dst[0] = (_Float16)a[0]; dst[1] = (_Float16)a[1];
                dst[2] = (_Float16)a[2]; dst[3] = (_Float16)a[3];
            }
        }
        // ---- stage B tile as N x K in LDS ----
        if constexpr (BT) {                      // B is N x K row-major (W for x@W.T)
            constexpr int RP = THREADS / (BK / 4);
            const int lr = tid / (BK / 4);
            const int lc = (tid % (BK / 4)) * 4;
            #pragma unroll 4
            for (int r = 0; r < BN; r += RP) {
                const float* src = B + (long long)(n0 + r + lr) * ldb + kt + lc;
                __builtin_prefetch(src + BK, 0, 1);
                v4f b = *(const v4f*)src;
                _Float16* dst = &Bs[(r + lr) * LDK + lc];
                dst[0] = (_Float16)b[0]; dst[1] = (_Float16)b[1];
                dst[2] = (_Float16)b[2]; dst[3] = (_Float16)b[3];
            }
        } else {                                 // B is K x N row-major -> transpose into LDS
            constexpr int CH = BN / 4;           // float4 chunks per k-row
            constexpr int RP = THREADS / CH;     // k-rows per pass
            const int kr = tid / CH;
            const int nc = (tid % CH) * 4;
            #pragma unroll 4
            for (int r = 0; r < BK; r += RP) {
                const float* src = B + (long long)(kt + r + kr) * ldb + n0 + nc;
                v4f b = *(const v4f*)src;
                #pragma unroll
                for (int j = 0; j < 4; ++j)
                    Bs[(nc + j) * LDK + (r + kr)] = (_Float16)b[j];
            }
        }
        __syncthreads();

        // ---- compute: two 16x16x32 k-steps per LDS tile ----
        #pragma unroll
        for (int ks = 0; ks < BK; ks += 32) {
            FragAB af[MI], bf[NI];
            #pragma unroll
            for (int mi = 0; mi < MI; ++mi) {
                // A 16-bit layout: half group g holds K = ks+g*8..+7 and ks+16+g*8..+7
                const _Float16* ap = &As[(wrow * WM + mi * 16 + l16) * LDK + ks + g * 8];
                af[mi].h[0] = *(const v8h*)(ap);
                af[mi].h[1] = *(const v8h*)(ap + 16);
            }
            #pragma unroll
            for (int ni = 0; ni < NI; ++ni) {
                // B 16-bit layout: half group g holds K = ks+g*8..+7 and ks+16+g*8..+7
                const _Float16* bp = &Bs[(wcol * WN + ni * 16 + l16) * LDK + ks + g * 8];
                bf[ni].h[0] = *(const v8h*)(bp);
                bf[ni].h[1] = *(const v8h*)(bp + 16);
            }
            #pragma unroll
            for (int mi = 0; mi < MI; ++mi)
                #pragma unroll
                for (int ni = 0; ni < NI; ++ni)
                    acc[mi][ni] = __builtin_amdgcn_wmma_f32_16x16x32_f16(
                        false, af[mi].v, false, bf[ni].v,
                        (short)0, acc[mi][ni], false, false);
        }
        __syncthreads();
    }

    // ---- epilogue + store (C layout: elem i -> row base+g*8+i, col l16) ----
    #pragma unroll
    for (int mi = 0; mi < MI; ++mi) {
        const int mbase = m0 + wrow * WM + mi * 16 + g * 8;
        #pragma unroll
        for (int ni = 0; ni < NI; ++ni) {
            const int col = n0 + wcol * WN + ni * 16 + l16;
            float bv = 0.f;
            if constexpr (EPI >= 1) bv = bias[col];
            #pragma unroll
            for (int i = 0; i < 8; ++i) {
                float v = acc[mi][ni][i] + bv;
                if constexpr (EPI == 2)
                    v = 0.5f * v * (1.0f + erff(v * 0.70710678118654752f));
                C[(long long)(mbase + i) * ldc + col] = v;
            }
        }
    }
}

// =====================================================================
// Embedding gather * sqrt(D) + sinusoidal positional encoding
// =====================================================================
__global__ __launch_bounds__(256) void embed_kernel(
    float* __restrict__ x, const int* __restrict__ ids, const float* __restrict__ emb)
{
    const int t = blockIdx.x;                 // token = b*S + s
    const int s = t % SDIM;
    const long long id = ids[t];
    const float sq = 32.0f;                   // sqrt(1024)
    for (int d = threadIdx.x; d < DDIM; d += blockDim.x) {
        const int i2 = (d >> 1) * 2;
        const float div = expf(-(float)i2 * (9.210340371976184f / (float)DDIM));
        const float ang = (float)s * div;
        const float pe = (d & 1) ? cosf(ang) : sinf(ang);
        x[(long long)t * DDIM + d] = emb[id * DDIM + d] * sq + pe;
    }
}

// =====================================================================
// out = LayerNorm(x + r) * g + b       (one block per token, D=1024)
// =====================================================================
__global__ __launch_bounds__(256) void add_ln_kernel(
    float* __restrict__ out, const float* __restrict__ x, const float* __restrict__ r,
    const float* __restrict__ g, const float* __restrict__ b)
{
    __shared__ float sh[256];
    const int t = blockIdx.x, tid = threadIdx.x;
    const long long base = (long long)t * DDIM;
    float v[4]; float s = 0.f;
    #pragma unroll
    for (int j = 0; j < 4; ++j) { int d = tid + j * 256; v[j] = x[base + d] + r[base + d]; s += v[j]; }
    sh[tid] = s; __syncthreads();
    for (int st = 128; st > 0; st >>= 1) { if (tid < st) sh[tid] += sh[tid + st]; __syncthreads(); }
    const float mean = sh[0] * (1.0f / DDIM); __syncthreads();
    float sq = 0.f;
    #pragma unroll
    for (int j = 0; j < 4; ++j) { float d0 = v[j] - mean; sq += d0 * d0; }
    sh[tid] = sq; __syncthreads();
    for (int st = 128; st > 0; st >>= 1) { if (tid < st) sh[tid] += sh[tid + st]; __syncthreads(); }
    const float inv = rsqrtf(sh[0] * (1.0f / DDIM) + 1e-5f);
    #pragma unroll
    for (int j = 0; j < 4; ++j) {
        int d = tid + j * 256;
        out[base + d] = (v[j] - mean) * inv * g[d] + b[d];
    }
}

// =====================================================================
// acc += comb[t,e] * ( LN(y + x) * g + b )   (ExpertLayer residual LN)
// =====================================================================
__global__ __launch_bounds__(256) void expert_ln_combine_kernel(
    float* __restrict__ acc, const float* __restrict__ y, const float* __restrict__ x,
    const float* __restrict__ g, const float* __restrict__ b,
    const float* __restrict__ comb, int e)
{
    __shared__ float sh[256];
    const int t = blockIdx.x, tid = threadIdx.x;
    const long long base = (long long)t * DDIM;
    float v[4]; float s = 0.f;
    #pragma unroll
    for (int j = 0; j < 4; ++j) { int d = tid + j * 256; v[j] = y[base + d] + x[base + d]; s += v[j]; }
    sh[tid] = s; __syncthreads();
    for (int st = 128; st > 0; st >>= 1) { if (tid < st) sh[tid] += sh[tid + st]; __syncthreads(); }
    const float mean = sh[0] * (1.0f / DDIM); __syncthreads();
    float sq = 0.f;
    #pragma unroll
    for (int j = 0; j < 4; ++j) { float d0 = v[j] - mean; sq += d0 * d0; }
    sh[tid] = sq; __syncthreads();
    for (int st = 128; st > 0; st >>= 1) { if (tid < st) sh[tid] += sh[tid + st]; __syncthreads(); }
    const float inv = rsqrtf(sh[0] * (1.0f / DDIM) + 1e-5f);
    const float w = comb[(long long)t * EEXP + e];
    if (w != 0.f) {
        #pragma unroll
        for (int j = 0; j < 4; ++j) {
            int d = tid + j * 256;
            acc[base + d] += w * ((v[j] - mean) * inv * g[d] + b[d]);
        }
    }
}

// =====================================================================
// Row softmax (length S=1024) with pre-scale (1/sqrt(hd))
// =====================================================================
__global__ __launch_bounds__(256) void softmax_kernel(float* __restrict__ s, float scale)
{
    __shared__ float sh[256];
    const long long base = (long long)blockIdx.x * SDIM;
    const int tid = threadIdx.x;
    float v[4]; float mx = -3.4e38f;
    #pragma unroll
    for (int j = 0; j < 4; ++j) { v[j] = s[base + tid + j * 256] * scale; mx = fmaxf(mx, v[j]); }
    sh[tid] = mx; __syncthreads();
    for (int st = 128; st > 0; st >>= 1) { if (tid < st) sh[tid] = fmaxf(sh[tid], sh[tid + st]); __syncthreads(); }
    mx = sh[0]; __syncthreads();
    float sum = 0.f;
    #pragma unroll
    for (int j = 0; j < 4; ++j) { v[j] = expf(v[j] - mx); sum += v[j]; }
    sh[tid] = sum; __syncthreads();
    for (int st = 128; st > 0; st >>= 1) { if (tid < st) sh[tid] += sh[tid + st]; __syncthreads(); }
    const float inv = 1.0f / sh[0];
    #pragma unroll
    for (int j = 0; j < 4; ++j) s[base + tid + j * 256] = v[j] * inv;
}

// =====================================================================
// Router: logits -> softmax -> top-2 renorm -> combine weights comb[t,E]
// one block per token; expert e = wave (8 waves of 32 lanes)
// =====================================================================
__global__ __launch_bounds__(256) void router_kernel(
    const float* __restrict__ x, const float* __restrict__ rw,
    const float* __restrict__ rb, float* __restrict__ comb)
{
    __shared__ float logits[EEXP];
    const int t = blockIdx.x;
    const int e = threadIdx.x >> 5, lane = threadIdx.x & 31;
    const float* xr = x + (long long)t * DDIM;
    const float* wr = rw + (long long)e * DDIM;
    float s = 0.f;
    for (int d = lane; d < DDIM; d += 32) s += xr[d] * wr[d];
    #pragma unroll
    for (int o = 16; o >= 1; o >>= 1) s += __shfl_xor(s, o, 32);
    if (lane == 0) logits[e] = s + rb[e];
    __syncthreads();
    if (threadIdx.x == 0) {
        float mx = logits[0];
        for (int i = 1; i < EEXP; ++i) mx = fmaxf(mx, logits[i]);
        float p[EEXP]; float sum = 0.f;
        for (int i = 0; i < EEXP; ++i) { p[i] = expf(logits[i] - mx); sum += p[i]; }
        for (int i = 0; i < EEXP; ++i) p[i] /= sum;
        int i0 = 0;
        for (int i = 1; i < EEXP; ++i) if (p[i] > p[i0]) i0 = i;
        int i1 = (i0 == 0) ? 1 : 0;
        for (int i = 0; i < EEXP; ++i) if (i != i0 && p[i] > p[i1]) i1 = i;
        const float w = p[i0] + p[i1];
        for (int i = 0; i < EEXP; ++i)
            comb[(long long)t * EEXP + i] = (i == i0) ? p[i0] / w : (i == i1) ? p[i1] / w : 0.f;
    }
}

// =====================================================================
// Host orchestration
// =====================================================================
extern "C" void kernel_launch(void* const* d_in, const int* in_sizes, int n_in,
                              void* d_out, int out_size, void* d_ws, size_t ws_size,
                              hipStream_t stream)
{
    (void)in_sizes; (void)n_in; (void)out_size; (void)ws_size;
    const int*   ids   = (const int*)  d_in[0];
    const float* emb   = (const float*)d_in[1];
    const float* qkv_w = (const float*)d_in[2];
    const float* qkv_b = (const float*)d_in[3];
    const float* out_w = (const float*)d_in[4];
    const float* out_b = (const float*)d_in[5];
    const float* ln1_g = (const float*)d_in[6];
    const float* ln1_b = (const float*)d_in[7];
    const float* rw    = (const float*)d_in[8];
    const float* rb    = (const float*)d_in[9];
    const float* fc1_w = (const float*)d_in[10];
    const float* fc1_b = (const float*)d_in[11];
    const float* fc2_w = (const float*)d_in[12];
    const float* fc2_b = (const float*)d_in[13];
    const float* eln_g = (const float*)d_in[14];
    const float* eln_b = (const float*)d_in[15];
    const float* ln2_g = (const float*)d_in[16];
    const float* ln2_b = (const float*)d_in[17];
    const float* pw    = (const float*)d_in[18];
    const float* pb    = (const float*)d_in[19];

    const int BS = BDIM * SDIM;                                  // 2048 tokens
    float* ws = (float*)d_ws;
    float* x      = ws; ws += (long long)BS * DDIM;              // 8 MB
    float* qkv    = ws; ws += (long long)BS * 3 * DDIM;          // 24 MB
    float* scores = ws; ws += (long long)BDIM * NHEAD * SDIM * SDIM; // 128 MB
    float* ctx    = ws; ws += (long long)BS * DDIM;              // 8 MB
    float* tmp    = ws; ws += (long long)BS * DDIM;              // 8 MB
    float* hbuf   = ws; ws += (long long)BS * HDIM;              // 32 MB
    float* ybuf   = ws; ws += (long long)BS * DDIM;              // 8 MB
    float* moe    = ws; ws += (long long)BS * DDIM;              // 8 MB
    float* comb   = ws; ws += (long long)BS * EEXP;

    embed_kernel<<<BS, 256, 0, stream>>>(x, ids, emb);

    for (int l = 0; l < NLAY; ++l) {
        // qkv = x @ qkv_w[l].T + qkv_b[l]         [BS, 3D]
        gemm_f16_wmma<128,128,64,2,2,true,1><<<dim3(3*DDIM/128, BS/128, 1), 128, 0, stream>>>(
            x, qkv_w + (long long)l*3*DDIM*DDIM, qkv, qkv_b + (long long)l*3*DDIM,
            BS, 3*DDIM, DDIM, DDIM, DDIM, 3*DDIM, 1, 0,0, 0,0, 0,0);

        // scores[b,h] = q_bh @ k_bh.T             z = b*NH+h, K = hd
        gemm_f16_wmma<128,128,64,2,2,true,0><<<dim3(SDIM/128, SDIM/128, BDIM*NHEAD), 128, 0, stream>>>(
            qkv, qkv + DDIM, scores, nullptr,
            SDIM, SDIM, HDHD, 3*DDIM, 3*DDIM, SDIM,
            NHEAD, (long long)SDIM*3*DDIM, HDHD,
                   (long long)SDIM*3*DDIM, HDHD,
                   (long long)NHEAD*SDIM*SDIM, (long long)SDIM*SDIM);

        softmax_kernel<<<BDIM*NHEAD*SDIM, 256, 0, stream>>>(scores, 0.125f); // 1/sqrt(64)

        // ctx[b,:,h*64:] = attn @ v_bh            B is KxN (v is S x hd)
        gemm_f16_wmma<128,64,64,2,2,false,0><<<dim3(1, SDIM/128, BDIM*NHEAD), 128, 0, stream>>>(
            scores, qkv + 2*DDIM, ctx, nullptr,
            SDIM, HDHD, SDIM, SDIM, 3*DDIM, DDIM,
            NHEAD, (long long)NHEAD*SDIM*SDIM, (long long)SDIM*SDIM,
                   (long long)SDIM*3*DDIM, HDHD,
                   (long long)SDIM*DDIM, HDHD);

        // tmp = ctx @ out_w[l].T + out_b[l]
        gemm_f16_wmma<128,128,64,2,2,true,1><<<dim3(DDIM/128, BS/128, 1), 128, 0, stream>>>(
            ctx, out_w + (long long)l*DDIM*DDIM, tmp, out_b + (long long)l*DDIM,
            BS, DDIM, DDIM, DDIM, DDIM, DDIM, 1, 0,0, 0,0, 0,0);

        add_ln_kernel<<<BS, 256, 0, stream>>>(x, x, tmp, ln1_g + l*DDIM, ln1_b + l*DDIM);

        // MoE
        router_kernel<<<BS, 256, 0, stream>>>(x, rw + (long long)l*EEXP*DDIM, rb + l*EEXP, comb);
        hipMemsetAsync(moe, 0, (size_t)BS * DDIM * sizeof(float), stream);
        for (int e = 0; e < EEXP; ++e) {
            const long long we = (long long)(l * EEXP + e);
            // h = gelu(x @ fc1_w[l,e].T + fc1_b[l,e])   [BS, H]
            gemm_f16_wmma<128,128,64,2,2,true,2><<<dim3(HDIM/128, BS/128, 1), 128, 0, stream>>>(
                x, fc1_w + we*HDIM*DDIM, hbuf, fc1_b + we*HDIM,
                BS, HDIM, DDIM, DDIM, DDIM, HDIM, 1, 0,0, 0,0, 0,0);
            // y = h @ fc2_w[l,e].T + fc2_b[l,e]          [BS, D]
            gemm_f16_wmma<128,128,64,2,2,true,1><<<dim3(DDIM/128, BS/128, 1), 128, 0, stream>>>(
                hbuf, fc2_w + we*DDIM*HDIM, ybuf, fc2_b + we*DDIM,
                BS, DDIM, HDIM, HDIM, HDIM, DDIM, 1, 0,0, 0,0, 0,0);
            // moe += comb[:,e] * LN(y + x)
            expert_ln_combine_kernel<<<BS, 256, 0, stream>>>(
                moe, ybuf, x, eln_g + we*DDIM, eln_b + we*DDIM, comb, e);
        }
        add_ln_kernel<<<BS, 256, 0, stream>>>(x, x, moe, ln2_g + l*DDIM, ln2_b + l*DDIM);
    }

    // logits = x @ proj_w.T + proj_b              [BS, V]
    gemm_f16_wmma<128,128,64,2,2,true,1><<<dim3(VVOC/128, BS/128, 1), 128, 0, stream>>>(
        x, pw, (float*)d_out, pb,
        BS, VVOC, DDIM, DDIM, DDIM, VVOC, 1, 0,0, 0,0, 0,0);
}
